// AttnLayerBU_47837345743235
// MI455X (gfx1250) — matrix operand order repaired and checked by reading, not compiled, not run on hardware
//
#include <hip/hip_runtime.h>

#define HD    32
#define NHEAD 8
#define DIMC  256

typedef __attribute__((ext_vector_type(16))) _Float16 v16h;
typedef __attribute__((ext_vector_type(8)))  _Float16 v8h;
typedef __attribute__((ext_vector_type(8)))  float    v8f;

union F16Frag { v16h v; v8h h[2]; };

__device__ __forceinline__ v8f wmma_f16f32(v16h a, v16h b, v8f c) {
  // D = A(16x32,f16) x B(32x16,f16) + C(16x16,f32)
  return __builtin_amdgcn_wmma_f32_16x16x32_f16(false, a, false, b, (short)0, c,
                                                false, false);
}

__global__ void cvt_f32_to_f16(const float* __restrict__ s, _Float16* __restrict__ d,
                               long n) {
  long i = (long)blockIdx.x * blockDim.x + threadIdx.x;
  if (i < n) d[i] = (_Float16)s[i];
}

// Out[b,h,l,d] = scale * (X[b,l,:] @ Wt[n,:] + bias[n]),  n = h*32+d
// X: f16 [Bb, L, 256] dense.  Out: f16 [Bb, NHEAD, Lpad, 32], pad rows zeroed.
// One wave computes a 16x64 output tile; K-loop is double-buffered so next
// chunk's fragments load while the current chunk's 4 WMMAs execute.
__global__ void proj_gemm(const _Float16* __restrict__ X, const _Float16* __restrict__ Wt,
                          const float* __restrict__ bias, float scale,
                          _Float16* __restrict__ Out, int Bb, int L, int Lpad) {
  const int lane = threadIdx.x & 31;
  const int l16  = lane & 15;
  const int half = lane >> 4;
  const long w = (long)blockIdx.x * (blockDim.x >> 5) + (threadIdx.x >> 5);
  const int ltiles = Lpad >> 4;
  const long total = (long)Bb * ltiles * 4;             // 4 n-tiles of 64 cols
  if (w >= total) return;
  const int nt4 = (int)(w % 4);
  const int lt  = (int)((w / 4) % ltiles);
  const long b  = w / (4L * ltiles);

  const int mrow = lt * 16 + l16;                       // A-fragment row
  const _Float16* arow = X + (b * L + (mrow < L ? mrow : 0)) * DIMC;
  const int nbase = nt4 * 64;
  const _Float16* brow[4];
#pragma unroll
  for (int j = 0; j < 4; ++j)
    brow[j] = Wt + (long)(nbase + j * 16 + l16) * DIMC + half * 16;
  const int ak0 = half * 16;
  const int ak1 = half * 16 + 8;

  v8f acc[4] = {};
  F16Frag a[2], bf[2][4];
  a[0].h[0] = *(const v8h*)(arow + ak0);
  a[0].h[1] = *(const v8h*)(arow + ak1);
#pragma unroll
  for (int j = 0; j < 4; ++j) {
    bf[0][j].h[0] = *(const v8h*)(brow[j]);
    bf[0][j].h[1] = *(const v8h*)(brow[j] + 8);
  }
#pragma unroll
  for (int kc = 0; kc < DIMC; kc += 32) {
    const int cur = (kc >> 5) & 1;
    const int nxt = cur ^ 1;
    if (kc + 32 < DIMC) {
      a[nxt].h[0] = *(const v8h*)(arow + kc + 32 + ak0);
      a[nxt].h[1] = *(const v8h*)(arow + kc + 32 + ak1);
#pragma unroll
      for (int j = 0; j < 4; ++j) {
        bf[nxt][j].h[0] = *(const v8h*)(brow[j] + kc + 32);
        bf[nxt][j].h[1] = *(const v8h*)(brow[j] + kc + 40);
      }
    }
#pragma unroll
    for (int j = 0; j < 4; ++j)
      acc[j] = wmma_f16f32(a[cur].v, bf[cur][j].v, acc[j]);
  }
#pragma unroll
  for (int j = 0; j < 4; ++j) {
    const int nglob = nbase + j * 16 + l16;
    const float bn = bias[nglob];
    const int hh = nglob >> 5;
    const int dd = nglob & 31;
#pragma unroll
    for (int r = 0; r < 8; ++r) {
      const int l = lt * 16 + r + 8 * half;             // C layout: M = r + 8*half
      const float v = (l < L) ? scale * (acc[j][r] + bn) : 0.0f;
      Out[((b * NHEAD + hh) * (long)Lpad + l) * HD + dd] = (_Float16)v;
    }
  }
}

// One wave per (batch, head, 16-query tile).  Qh/Kh/Vh: f16 [B,H,Lpad,32].
// O: f16 [B*Lq, 256] token-major (head-concatenated), ready for out-projection.
__global__ void attn_kernel(const _Float16* __restrict__ Qh, const _Float16* __restrict__ Kh,
                            const _Float16* __restrict__ Vh, _Float16* __restrict__ O,
                            int Bb, int Lq, int LqPad, int Lk, int LkPad) {
  __shared__ __align__(16) _Float16 Pl[4][16][128];     // per-wave P staging
  const int lane  = threadIdx.x & 31;
  const int l16   = lane & 15;
  const int half  = lane >> 4;
  const int wslot = threadIdx.x >> 5;
  const int qtiles = LqPad >> 4;
  const long total = (long)Bb * NHEAD * qtiles;
  long w = (long)blockIdx.x * 4 + wslot;
  if (w >= total) w = total - 1;                        // clamp: keep waves alive for barrier
  const int qt = (int)(w % qtiles);
  const int hh = (int)((w / qtiles) % NHEAD);
  const long b = w / ((long)qtiles * NHEAD);
  const long bh = b * NHEAD + hh;

  const int ak0 = half * 16, ak1 = half * 16 + 8;

  // Q fragment (A-layout): 16 query rows x 32 head dims
  const _Float16* qrow = Qh + (bh * LqPad + qt * 16 + l16) * HD;
  F16Frag aq;
  aq.h[0] = *(const v8h*)(qrow + ak0);
  aq.h[1] = *(const v8h*)(qrow + ak1);

  // Scores: S[16, LkPad] = Q x K^T, one WMMA per 16 keys (K dim = head_dim = 32)
  const int ntk = LkPad >> 4;                           // <= 8
  v8f sc[8];
  for (int t = 0; t < ntk; ++t) {
    const _Float16* krow = Kh + (bh * LkPad + t * 16 + l16) * HD + half * 16;
    F16Frag bk;
    bk.h[0] = *(const v8h*)(krow);
    bk.h[1] = *(const v8h*)(krow + 8);
    v8f z = {};
    sc[t] = wmma_f16f32(aq.v, bk.v, z);
  }
  // Mask padded keys
  for (int t = 0; t < ntk; ++t) {
    if (t * 16 + l16 >= Lk) {
#pragma unroll
      for (int r = 0; r < 8; ++r) sc[t][r] = -1e30f;
    }
  }
  // Row softmax: row M = r + 8*half lives across the 16 lanes of this half-wave
#pragma unroll
  for (int r = 0; r < 8; ++r) {
    float mx = -1e30f;
    for (int t = 0; t < ntk; ++t) mx = fmaxf(mx, sc[t][r]);
    for (int m = 1; m <= 8; m <<= 1) mx = fmaxf(mx, __shfl_xor(mx, m, 32));
    float s = 0.f;
    for (int t = 0; t < ntk; ++t) { float e = __expf(sc[t][r] - mx); sc[t][r] = e; s += e; }
    for (int m = 1; m <= 8; m <<= 1) s += __shfl_xor(s, m, 32);
    const float inv = 1.0f / s;
    for (int t = 0; t < ntk; ++t)
      Pl[wslot][r + 8 * half][t * 16 + l16] = (_Float16)(sc[t][r] * inv);
  }
  __syncthreads();

  // O = P(16 x LkPad) @ V(LkPad x 32): two 16-wide N tiles, K chunks of 32 keys
#pragma unroll
  for (int t2 = 0; t2 < 2; ++t2) {
    v8f acc = {};
    for (int kc = 0; kc < LkPad; kc += 32) {
      F16Frag ap, bv;
      ap.h[0] = *(const v8h*)(&Pl[wslot][l16][kc + ak0]);
      ap.h[1] = *(const v8h*)(&Pl[wslot][l16][kc + ak1]);
      const _Float16* vcol = Vh + (bh * LkPad + kc + half * 16) * HD + t2 * 16 + l16;
#pragma unroll
      for (int j = 0; j < 16; ++j) bv.v[j] = vcol[(long)j * HD];
      acc = wmma_f16f32(ap.v, bv.v, acc);
    }
#pragma unroll
    for (int r = 0; r < 8; ++r) {
      const int l = qt * 16 + r + 8 * half;
      if (l < Lq)
        O[(b * Lq + l) * (long)DIMC + hh * HD + t2 * 16 + l16] = (_Float16)acc[r];
    }
  }
}

// Out[t,n] (+)= softmax(wvec)[widx] * (A[t,:] @ Wt[n,:] + bias[n]),  A f16, Out f32
// One wave computes a 16x64 output tile; double-buffered K-loop.
__global__ void out_gemm(const _Float16* __restrict__ A, const _Float16* __restrict__ Wt,
                         const float* __restrict__ bias, const float* __restrict__ wvec,
                         int widx, int wcount, float* __restrict__ Out, long T, int accum) {
  const int lane = threadIdx.x & 31;
  const int l16  = lane & 15;
  const int half = lane >> 4;
  const long w = (long)blockIdx.x * (blockDim.x >> 5) + (threadIdx.x >> 5);
  const long ttiles = (T + 15) >> 4;
  const long total = ttiles * 4;                        // 4 n-tiles of 64 cols
  if (w >= total) return;
  const int nt4 = (int)(w % 4);
  const long tt = w / 4;

  const long trow = tt * 16 + l16;
  const _Float16* arow = A + (trow < T ? trow : T - 1) * DIMC;
  const int nbase = nt4 * 64;
  const _Float16* brow[4];
#pragma unroll
  for (int j = 0; j < 4; ++j)
    brow[j] = Wt + (long)(nbase + j * 16 + l16) * DIMC + half * 16;
  const int ak0 = half * 16, ak1 = half * 16 + 8;

  v8f acc[4] = {};
  F16Frag a[2], bf[2][4];
  a[0].h[0] = *(const v8h*)(arow + ak0);
  a[0].h[1] = *(const v8h*)(arow + ak1);
#pragma unroll
  for (int j = 0; j < 4; ++j) {
    bf[0][j].h[0] = *(const v8h*)(brow[j]);
    bf[0][j].h[1] = *(const v8h*)(brow[j] + 8);
  }
#pragma unroll
  for (int kc = 0; kc < DIMC; kc += 32) {
    const int cur = (kc >> 5) & 1;
    const int nxt = cur ^ 1;
    if (kc + 32 < DIMC) {
      a[nxt].h[0] = *(const v8h*)(arow + kc + 32 + ak0);
      a[nxt].h[1] = *(const v8h*)(arow + kc + 32 + ak1);
#pragma unroll
      for (int j = 0; j < 4; ++j) {
        bf[nxt][j].h[0] = *(const v8h*)(brow[j] + kc + 32);
        bf[nxt][j].h[1] = *(const v8h*)(brow[j] + kc + 40);
      }
    }
#pragma unroll
    for (int j = 0; j < 4; ++j)
      acc[j] = wmma_f16f32(a[cur].v, bf[cur][j].v, acc[j]);
  }
  // softmax-selected blend weight (2 or 3 elements)
  float mx = wvec[0];
  for (int i = 1; i < wcount; ++i) mx = fmaxf(mx, wvec[i]);
  float s = 0.f, num = 0.f;
  for (int i = 0; i < wcount; ++i) {
    float e = __expf(wvec[i] - mx);
    s += e;
    if (i == widx) num = e;
  }
  const float wsel = num / s;
#pragma unroll
  for (int j = 0; j < 4; ++j) {
    const int nglob = nbase + j * 16 + l16;
    const float bn = bias[nglob];
#pragma unroll
    for (int r = 0; r < 8; ++r) {
      const long t = tt * 16 + r + 8 * half;
      if (t < T) {
        const float v = wsel * (acc[j][r] + bn);
        const long idx = t * DIMC + nglob;
        Out[idx] = accum ? (Out[idx] + v) : v;
      }
    }
  }
}

extern "C" void kernel_launch(void* const* d_in, const int* in_sizes, int n_in,
                              void* d_out, int out_size, void* d_ws, size_t ws_size,
                              hipStream_t stream) {
  (void)in_sizes; (void)n_in; (void)out_size; (void)ws_size;
  const float* leaf   = (const float*)d_in[0];
  const float* inner  = (const float*)d_in[1];
  const float* root   = (const float*)d_in[2];
  const float* w_in   = (const float*)d_in[3];
  const float* b_in   = (const float*)d_in[4];
  const float* w_out  = (const float*)d_in[5];
  const float* b_out  = (const float*)d_in[6];
  const float* wleaf  = (const float*)d_in[7];
  const float* winner = (const float*)d_in[8];
  const float* wroot  = (const float*)d_in[9];
  float* out = (float*)d_out;
  char* ws = (char*)d_ws;

  size_t off = 0;
  auto take = [&](size_t bytes) -> void* {
    void* p = ws + off;
    off += (bytes + 255) & ~(size_t)255;
    return p;
  };
  const long N_LEAF  = 2048L * 64;   // leaf tokens
  const long N_INNER = 2048L;        // 16*128
  const long N_ROOT  = 256L;         // 16*16
  _Float16* leaf16  = (_Float16*)take(N_LEAF  * DIMC * 2);
  _Float16* inner16 = (_Float16*)take(N_INNER * DIMC * 2);
  _Float16* root16  = (_Float16*)take(N_ROOT  * DIMC * 2);
  _Float16* win16   = (_Float16*)take(7L * 768 * DIMC * 2);
  _Float16* wout16  = (_Float16*)take(7L * DIMC * DIMC * 2);
  _Float16* Qh   = (_Float16*)take(131072L * DIMC * 2);
  _Float16* Kh   = (_Float16*)take(131072L * DIMC * 2);
  _Float16* Vh   = (_Float16*)take(131072L * DIMC * 2);
  _Float16* Obuf = (_Float16*)take(131072L * DIMC * 2);

  auto cvt = [&](const float* s, _Float16* d, long n) {
    unsigned blocks = (unsigned)((n + 255) / 256);
    cvt_f32_to_f16<<<dim3(blocks), dim3(256), 0, stream>>>(s, d, n);
  };
  cvt(leaf,  leaf16,  N_LEAF  * DIMC);
  cvt(inner, inner16, N_INNER * DIMC);
  cvt(root,  root16,  N_ROOT  * DIMC);
  cvt(w_in,  win16,   7L * 768 * DIMC);
  cvt(w_out, wout16,  7L * DIMC * DIMC);

  auto proj = [&](const _Float16* X, const _Float16* Wt, const float* bias, float scale,
                  _Float16* Out, int Bb, int L, int Lpad) {
    long waves = (long)Bb * (Lpad / 16) * 4;
    unsigned blocks = (unsigned)((waves + 7) / 8);
    proj_gemm<<<dim3(blocks), dim3(256), 0, stream>>>(X, Wt, bias, scale, Out, Bb, L, Lpad);
  };

  struct Cfg {
    const _Float16* xq; int Bq, Lq, LqPad;
    const _Float16* xkv; int Lk, LkPad;
    int wi, wo;
    const float* wvec; int wsel, wcount;
    long out_off; int accum;
  };
  const long OFF_INNER = 2048L * 64 * 256;
  const long OFF_ROOT  = OFF_INNER + 2048L * 256;
  // Note: reference uses w_in[5]/w_out[5] for BOTH root_s and root_c.
  const Cfg calls[7] = {
    { leaf16,  2048,  64,  64, inner16,   1,  32, 0, 0, wleaf,  0, 2, 0,         0 },
    { leaf16,  2048,  64,  64, leaf16,   64,  64, 1, 1, wleaf,  1, 2, 0,         1 },
    { inner16,   16, 128, 128, root16,   16,  32, 2, 2, winner, 0, 3, OFF_INNER, 0 },
    { inner16,   16, 128, 128, inner16, 128, 128, 3, 3, winner, 1, 3, OFF_INNER, 1 },
    { inner16, 2048,   1,  16, leaf16,   64,  64, 4, 4, winner, 2, 3, OFF_INNER, 1 },
    { root16,    16,  16,  16, root16,   16,  32, 5, 5, wroot,  0, 2, OFF_ROOT,  0 },
    { root16,    16,  16,  16, inner16, 128, 128, 5, 5, wroot,  1, 2, OFF_ROOT,  1 },
  };
  const float QSCALE = 0.17677669529663687f;  // 1/sqrt(32)

  for (int c = 0; c < 7; ++c) {
    const Cfg& g = calls[c];
    const _Float16* wt = win16 + (long)g.wi * 768 * DIMC;
    const float* bi = b_in + g.wi * 768;
    proj(g.xq,  wt,               bi,       QSCALE, Qh, g.Bq, g.Lq, g.LqPad);
    proj(g.xkv, wt + 256L * DIMC, bi + 256, 1.0f,   Kh, g.Bq, g.Lk, g.LkPad);
    proj(g.xkv, wt + 512L * DIMC, bi + 512, 1.0f,   Vh, g.Bq, g.Lk, g.LkPad);

    long awaves = (long)g.Bq * NHEAD * (g.LqPad / 16);
    attn_kernel<<<dim3((unsigned)((awaves + 3) / 4)), dim3(128), 0, stream>>>(
        Qh, Kh, Vh, Obuf, g.Bq, g.Lq, g.LqPad, g.Lk, g.LkPad);

    long T = (long)g.Bq * g.Lq;
    long owaves = ((T + 15) / 16) * 4;
    out_gemm<<<dim3((unsigned)((owaves + 7) / 8)), dim3(256), 0, stream>>>(
        Obuf, wout16 + (long)g.wo * DIMC * DIMC, b_out + g.wo * DIMC,
        g.wvec, g.wsel, g.wcount, out + g.out_off, T, g.accum);
  }
}
